// QwenAttention_10900626997987
// MI455X (gfx1250) — compile-verified, matrix operand order checked by reading, not run on hardware
//
#include <hip/hip_runtime.h>
#include <hip/hip_bf16.h>
#include <cstdint>
#include <cstddef>

#define B_   2
#define S_   2048
#define H_   896
#define NH_  14
#define NKV_ 2
#define HD_  64
#define GROUPS_ (NH_/NKV_)
#define M_   (B_*S_)            // 4096 tokens

typedef __bf16 bf16;
typedef __attribute__((ext_vector_type(8)))  bf16  v8bf;
typedef __attribute__((ext_vector_type(16))) bf16  v16bf;
typedef __attribute__((ext_vector_type(8)))  float v8f;

static __device__ inline bf16 f2bf(float f) {
    unsigned u = __builtin_bit_cast(unsigned, f);
    unsigned r = (u + 0x7FFFu + ((u >> 16) & 1u)) >> 16;
    unsigned short h = (unsigned short)r;
    return __builtin_bit_cast(bf16, h);
}

// Load a 16-bit WMMA fragment (16 striped-rows x 32 contiguous-K) from row-major
// memory. Per CDNA5 ISA (CDNA5 ISA): lanes 0-15 hold striped-index 0-15 with
// K = 0..7 (elems 0-7) and K = 16..23 (elems 8-15); lanes 16-31 hold the same
// striped indices with K = 8..15 and K = 24..31. Both runs are contiguous
// 16-byte loads.
static __device__ inline v16bf load_frag(const bf16* __restrict__ src, int ld) {
    int lane = threadIdx.x & 31;
    int row  = lane & 15;
    int half = lane >> 4;
    const v8bf lo = *reinterpret_cast<const v8bf*>(src + (size_t)row * ld + half * 8);
    const v8bf hi = *reinterpret_cast<const v8bf*>(src + (size_t)row * ld + 16 + half * 8);
    v16bf a;
#pragma unroll
    for (int i = 0; i < 8; ++i) { a[i] = lo[i]; a[i + 8] = hi[i]; }
    return a;
}

static __device__ inline v8f wmma_bf16(v16bf a, v16bf b, v8f c) {
    return __builtin_amdgcn_wmma_f32_16x16x32_bf16(false, a, false, b, (short)0, c,
                                                   false, false);
}

// ---------------------------------------------------------------------------
// Elementwise fp32 -> bf16
__global__ void __launch_bounds__(256) cvt_kernel(const float* __restrict__ in,
                                                  bf16* __restrict__ out, int n) {
    int i = blockIdx.x * blockDim.x + threadIdx.x;
    if (i < n) out[i] = f2bf(in[i]);
}

// Transpose + convert: W[K][N] fp32 -> WT[N][K] bf16 (B-fragments then load
// contiguously along K).
__global__ void __launch_bounds__(256) tcvt_kernel(const float* __restrict__ in,
                                                   bf16* __restrict__ out, int K, int N) {
    int i = blockIdx.x * blockDim.x + threadIdx.x;
    if (i < K * N) {
        int k = i / N, n = i - k * N;
        out[(size_t)n * K + k] = f2bf(in[i]);
    }
}

// ---------------------------------------------------------------------------
// GEMM: C[16x64 strip] = A[16xK] * WT^T, one wave per (m-tile, head) strip.
// All five fragment loads of a k-step are issued before the WMMA batch so the
// backend can use partial s_wait_loadcnt values and keep the XDL pipe fed.
// mode 0: +bias, RoPE, store Q  bf16 [B][NH][S][HD]
// mode 1: +bias, RoPE, store K  bf16 [B][NKV][S][HD]
// mode 2: +bias,       store V^T bf16 [B][NKV][HD][S]
// mode 3:              store fp32 [M][N]
__global__ void __launch_bounds__(128)
gemm_kernel(const bf16* __restrict__ A, const bf16* __restrict__ WT,
            const float* __restrict__ bias, bf16* __restrict__ out_bf,
            float* __restrict__ out_f, int K, int mode, int nheads) {
    const int wid   = threadIdx.x >> 5;
    const int strip = blockIdx.x * (blockDim.x >> 5) + wid;
    const int total = (M_ / 16) * nheads;
    if (strip >= total) return;

    const int mt   = strip / nheads;
    const int head = strip - mt * nheads;
    const int m0   = mt * 16;
    const int n0   = head * 64;
    const int lane = threadIdx.x & 31;
    const int col  = lane & 15;
    const int half = lane >> 4;

    v8f acc[4];
#pragma unroll
    for (int t = 0; t < 4; ++t)
#pragma unroll
        for (int r = 0; r < 8; ++r) acc[t][r] = 0.0f;

    for (int k0 = 0; k0 < K; k0 += 32) {
        if (k0 + 32 < K) {
            // pull the next k-tile toward the caches (global_prefetch_b8)
            __builtin_prefetch(A + (size_t)m0 * K + k0 + 32, 0, 1);
            __builtin_prefetch(WT + (size_t)n0 * K + k0 + 32, 0, 1);
        }
        // issue all loads first ...
        v16bf a = load_frag(A + (size_t)m0 * K + k0, K);
        v16bf bt[4];
#pragma unroll
        for (int t = 0; t < 4; ++t)
            bt[t] = load_frag(WT + (size_t)(n0 + t * 16) * K + k0, K);
        // ... then the WMMA batch
#pragma unroll
        for (int t = 0; t < 4; ++t)
            acc[t] = wmma_bf16(a, bt[t], acc[t]);
    }

    if (bias) {
#pragma unroll
        for (int t = 0; t < 4; ++t) {
            float bv = bias[n0 + t * 16 + col];
#pragma unroll
            for (int r = 0; r < 8; ++r) acc[t][r] += bv;
        }
    }

    if (mode == 0 || mode == 1) {
        // RoPE: d and d+32 are accumulators t and t+2 in the same lane.
        const float LOG_THETA_OVER_HALF = 0.43173470493638356f; // ln(1e6)/32
        const float freq0 = __expf(-(float)col * LOG_THETA_OVER_HALF);
        const float freq1 = __expf(-(float)(col + 16) * LOG_THETA_OVER_HALF);
#pragma unroll
        for (int r = 0; r < 8; ++r) {
            int row = r + half * 8;
            int tok = m0 + row;
            int b   = tok / S_;
            int s   = tok - b * S_;
            float a0 = (float)s * freq0, a1 = (float)s * freq1;
            float c0 = __cosf(a0), s0 = __sinf(a0);
            float c1 = __cosf(a1), s1 = __sinf(a1);
            float x0 = acc[0][r], x1 = acc[1][r], x2 = acc[2][r], x3 = acc[3][r];
            float o0 = x0 * c0 - x2 * s0;
            float o1 = x1 * c1 - x3 * s1;
            float o2 = x2 * c0 + x0 * s0;
            float o3 = x3 * c1 + x1 * s1;
            size_t base = (((size_t)b * nheads + head) * S_ + s) * HD_;
            out_bf[base + 0 + col]  = f2bf(o0);
            out_bf[base + 16 + col] = f2bf(o1);
            out_bf[base + 32 + col] = f2bf(o2);
            out_bf[base + 48 + col] = f2bf(o3);
        }
    } else if (mode == 2) {
#pragma unroll
        for (int r = 0; r < 8; ++r) {
            int row = r + half * 8;
            int tok = m0 + row;
            int b   = tok / S_;
            int s   = tok - b * S_;
            size_t base = ((size_t)b * nheads + head) * (size_t)HD_ * S_;
#pragma unroll
            for (int t = 0; t < 4; ++t)
                out_bf[base + (size_t)(t * 16 + col) * S_ + s] = f2bf(acc[t][r]);
        }
    } else {
#pragma unroll
        for (int r = 0; r < 8; ++r) {
            int row = r + half * 8;
            size_t base = (size_t)(m0 + row) * (size_t)(nheads * 64) + n0;
#pragma unroll
            for (int t = 0; t < 4; ++t) out_f[base + t * 16 + col] = acc[t][r];
        }
    }
}

// ---------------------------------------------------------------------------
// Flash attention: one wave per (b, h, 16-query tile). Online softmax over
// 64-key blocks (4 score tiles) to amortize the VALU/shuffle softmax cost over
// 16 back-to-back WMMAs per block. P transposed C-layout -> A-layout via a
// wave-private LDS slice (same-wave DS ops are in-order; no barrier needed).
__global__ void __launch_bounds__(128)
flash_kernel(const bf16* __restrict__ Q, const bf16* __restrict__ Kc,
             const bf16* __restrict__ Vt, bf16* __restrict__ AO) {
    __shared__ bf16 ldsP[4][16 * 64];

    const int wid = threadIdx.x >> 5;
    const int id  = blockIdx.x * (blockDim.x >> 5) + wid;  // 0..B*NH*128-1
    const int bh  = id >> 7;
    const int qt  = id & 127;
    const int b   = bh / NH_;
    const int h   = bh - b * NH_;
    const int hk  = h / GROUPS_;
    const int q0  = qt * 16;
    const int lane = threadIdx.x & 31;
    const int col  = lane & 15;
    const int half = lane >> 4;

    const bf16* Qh = Q  + ((size_t)b * NH_ + h) * (size_t)S_ * HD_;
    const bf16* Kh = Kc + ((size_t)b * NKV_ + hk) * (size_t)S_ * HD_;
    const bf16* Vh = Vt + ((size_t)b * NKV_ + hk) * (size_t)HD_ * S_;
    bf16* myP = &ldsP[wid][0];

    // Q tile A-fragments for d-chunks [0,32) and [32,64)
    v16bf aq0 = load_frag(Qh + (size_t)q0 * HD_ + 0, HD_);
    v16bf aq1 = load_frag(Qh + (size_t)q0 * HD_ + 32, HD_);

    float rm[8], rl[8];
    v8f o[4];
#pragma unroll
    for (int r = 0; r < 8; ++r) { rm[r] = -1e30f; rl[r] = 0.0f; }
#pragma unroll
    for (int t = 0; t < 4; ++t)
#pragma unroll
        for (int r = 0; r < 8; ++r) o[t][r] = 0.0f;

    for (int j0 = 0; j0 < q0 + 16; j0 += 64) {
        if (j0 + 64 < q0 + 16)
            __builtin_prefetch(Kh + (size_t)(j0 + 64) * HD_, 0, 1);

        // scores: four 16x16 tiles (keys j0..j0+63); loads first, WMMAs after
        v16bf bk[8];
#pragma unroll
        for (int kt = 0; kt < 4; ++kt) {
            bk[2 * kt + 0] = load_frag(Kh + (size_t)(j0 + kt * 16) * HD_ + 0, HD_);
            bk[2 * kt + 1] = load_frag(Kh + (size_t)(j0 + kt * 16) * HD_ + 32, HD_);
        }
        v8f sc[4];
#pragma unroll
        for (int kt = 0; kt < 4; ++kt) {
            v8f s0;
#pragma unroll
            for (int r = 0; r < 8; ++r) s0[r] = 0.0f;
            s0 = wmma_bf16(aq0, bk[2 * kt + 0], s0);
            s0 = wmma_bf16(aq1, bk[2 * kt + 1], s0);
            sc[kt] = s0;
        }
        // scale 1/sqrt(64) + analytic causal mask (matches -1e9 in reference)
#pragma unroll
        for (int kt = 0; kt < 4; ++kt)
#pragma unroll
            for (int r = 0; r < 8; ++r) {
                int key  = j0 + kt * 16 + col;
                int qrow = q0 + r + half * 8;
                float v = sc[kt][r] * 0.125f;
                sc[kt][r] = (key > qrow) ? -1.0e9f : v;
            }
        // online softmax row statistics (16-lane xor reductions within halves)
        float scale_r[8];
#pragma unroll
        for (int r = 0; r < 8; ++r) {
            float tm = fmaxf(fmaxf(sc[0][r], sc[1][r]), fmaxf(sc[2][r], sc[3][r]));
#pragma unroll
            for (int x = 1; x < 16; x <<= 1) tm = fmaxf(tm, __shfl_xor(tm, x, 32));
            float mnew = fmaxf(rm[r], tm);
            float sscl = __expf(rm[r] - mnew);
            rm[r] = mnew;
            float ps = 0.0f;
#pragma unroll
            for (int kt = 0; kt < 4; ++kt) {
                float p = __expf(sc[kt][r] - mnew);
                sc[kt][r] = p;
                ps += p;
            }
#pragma unroll
            for (int x = 1; x < 16; x <<= 1) ps += __shfl_xor(ps, x, 32);
            rl[r] = rl[r] * sscl + ps;
            scale_r[r] = sscl;
        }
#pragma unroll
        for (int t = 0; t < 4; ++t)
#pragma unroll
            for (int r = 0; r < 8; ++r) o[t][r] *= scale_r[r];

        // P (C-layout) -> LDS row-major [16][64] -> two A-fragments.
#pragma unroll
        for (int kt = 0; kt < 4; ++kt)
#pragma unroll
            for (int r = 0; r < 8; ++r)
                myP[(r + half * 8) * 64 + kt * 16 + col] = f2bf(sc[kt][r]);
        v16bf aP0 = load_frag(myP, 64);
        v16bf aP1 = load_frag(myP + 32, 64);

        // O += P @ V  (V stored transposed [HD][S]: contiguous along keys);
        // again all loads before the WMMA batch.
        v16bf bv[8];
#pragma unroll
        for (int t = 0; t < 4; ++t) {
            bv[2 * t + 0] = load_frag(Vh + (size_t)(t * 16) * S_ + j0, S_);
            bv[2 * t + 1] = load_frag(Vh + (size_t)(t * 16) * S_ + j0 + 32, S_);
        }
#pragma unroll
        for (int t = 0; t < 4; ++t) {
            o[t] = wmma_bf16(aP0, bv[2 * t + 0], o[t]);
            o[t] = wmma_bf16(aP1, bv[2 * t + 1], o[t]);
        }
    }

    // normalize and store attention output [B][S][NH*HD] bf16
#pragma unroll
    for (int r = 0; r < 8; ++r) {
        int row = r + half * 8;
        int s   = q0 + row;
        float inv = 1.0f / rl[r];
        size_t base = ((size_t)b * S_ + s) * H_ + (size_t)h * HD_;
#pragma unroll
        for (int t = 0; t < 4; ++t)
            AO[base + t * 16 + col] = f2bf(o[t][r] * inv);
    }
}

// ---------------------------------------------------------------------------
extern "C" void kernel_launch(void* const* d_in, const int* in_sizes, int n_in,
                              void* d_out, int out_size, void* d_ws, size_t ws_size,
                              hipStream_t stream) {
    const float* X  = (const float*)d_in[0];
    // d_in[1] = attention_mask (pure causal; applied analytically)
    const float* Wq = (const float*)d_in[2];
    const float* bq = (const float*)d_in[3];
    const float* Wk = (const float*)d_in[4];
    const float* bk = (const float*)d_in[5];
    const float* Wv = (const float*)d_in[6];
    const float* bv = (const float*)d_in[7];
    const float* Wo = (const float*)d_in[8];
    float* out = (float*)d_out;

    char* ws = (char*)d_ws;
    size_t off = 0;
    auto alloc = [&](size_t bytes) -> void* {
        void* p = ws + off;
        off = (off + bytes + 255) & ~(size_t)255;
        return p;
    };
    bf16* Xb  = (bf16*)alloc((size_t)M_ * H_ * 2);
    bf16* WqT = (bf16*)alloc((size_t)H_ * H_ * 2);            // [896][896]
    bf16* WkT = (bf16*)alloc((size_t)(NKV_ * HD_) * H_ * 2);  // [128][896]
    bf16* WvT = (bf16*)alloc((size_t)(NKV_ * HD_) * H_ * 2);
    bf16* WoT = (bf16*)alloc((size_t)H_ * H_ * 2);
    bf16* Qb  = (bf16*)alloc((size_t)B_ * NH_ * S_ * HD_ * 2);
    bf16* Kb  = (bf16*)alloc((size_t)B_ * NKV_ * S_ * HD_ * 2);
    bf16* Vtb = (bf16*)alloc((size_t)B_ * NKV_ * S_ * HD_ * 2);
    bf16* AO  = (bf16*)alloc((size_t)M_ * H_ * 2);
    (void)ws_size; (void)in_sizes; (void)n_in; (void)out_size;

    // Precision conversion / weight transposes
    int nX = M_ * H_;
    cvt_kernel<<<(nX + 255) / 256, 256, 0, stream>>>(X, Xb, nX);
    tcvt_kernel<<<(H_ * H_ + 255) / 256, 256, 0, stream>>>(Wq, WqT, H_, H_);
    tcvt_kernel<<<(H_ * NKV_ * HD_ + 255) / 256, 256, 0, stream>>>(Wk, WkT, H_, NKV_ * HD_);
    tcvt_kernel<<<(H_ * NKV_ * HD_ + 255) / 256, 256, 0, stream>>>(Wv, WvT, H_, NKV_ * HD_);
    tcvt_kernel<<<(H_ * H_ + 255) / 256, 256, 0, stream>>>(Wo, WoT, H_, H_);

    // QKV projections (+bias, +RoPE for Q/K, V stored transposed)
    // strips = (M/16)*heads ; 4 waves (strips) per 128-thread block
    gemm_kernel<<<(256 * NH_) / 4, 128, 0, stream>>>(Xb, WqT, bq, Qb, nullptr, H_, 0, NH_);
    gemm_kernel<<<(256 * NKV_) / 4, 128, 0, stream>>>(Xb, WkT, bk, Kb, nullptr, H_, 1, NKV_);
    gemm_kernel<<<(256 * NKV_) / 4, 128, 0, stream>>>(Xb, WvT, bv, Vtb, nullptr, H_, 2, NKV_);

    // Flash attention: B*NH*(S/16) = 3584 wave-tasks, 4 per block
    flash_kernel<<<(B_ * NH_ * (S_ / 16)) / 4, 128, 0, stream>>>(Qb, Kb, Vtb, AO);

    // Output projection -> fp32 d_out
    gemm_kernel<<<(256 * NH_) / 4, 128, 0, stream>>>(AO, WoT, nullptr, nullptr, out, H_, 3, NH_);
}
